// MambaBlock_12292196402033
// MI455X (gfx1250) — compile-verified, matrix operand order checked by reading, not run on hardware
//
#include <hip/hip_runtime.h>
#include <hip/hip_bf16.h>
#include <math.h>

// Mamba block, MI455X (gfx1250, wave32). Big GEMMs: 8-wave workgroups,
// 64x128 macro-tiles staged in LDS (async global->LDS when available),
// 2x2 v_wmma_f32_16x16x32_f16 register blocking per wave.

#define BATCH   4
#define SEQ     1024
#define DMODEL  512
#define DINNER  1024
#define DSTATE  16
#define DCONV   4
#define DTRANK  32
#define MROWS   (BATCH*SEQ)   // 4096 token rows

typedef __attribute__((ext_vector_type(16))) _Float16 v16h;
typedef __attribute__((ext_vector_type(8)))  _Float16 v8h;
typedef __attribute__((ext_vector_type(8)))  float    v8f;
typedef __attribute__((ext_vector_type(4)))  int      v4i_t;

#if defined(__HIP_DEVICE_COMPILE__) && \
    __has_builtin(__builtin_amdgcn_global_load_async_to_lds_b128) && \
    __has_builtin(__builtin_amdgcn_s_wait_asynccnt)
#define USE_ASYNC_LDS 1
typedef __attribute__((address_space(1))) v4i_t* gptr_t;   // global int4*
typedef __attribute__((address_space(3))) v4i_t* lptr_t;   // LDS int4*
#endif

__device__ __forceinline__ float sigmoidf_(float x) { return 1.f / (1.f + __expf(-x)); }

// Build one 16-bit WMMA fragment (A-layout, CDNA5 ISA 7.12.2):
// per lane: 8 halfs at p, 8 halfs at p+16 (K = g*8..g*8+7, 16+g*8..16+g*8+7)
__device__ __forceinline__ v16h ldfrag(const _Float16* p) {
  v8h lo = *(const v8h*)(p);
  v8h hi = *(const v8h*)(p + 16);
  return __builtin_shufflevector(lo, hi, 0,1,2,3,4,5,6,7,8,9,10,11,12,13,14,15);
}

__device__ __forceinline__ v8f wmma_step(v16h a, v16h b, v8f c) {
  return __builtin_amdgcn_wmma_f32_16x16x32_f16(false, a, false, b, (short)0, c,
                                                false, false);
}

// ---------------------------------------------------------------- converts
__global__ void cvt_f32_to_f16_k(const float* __restrict__ in,
                                 _Float16* __restrict__ out, int n) {
  int i = blockIdx.x * blockDim.x + threadIdx.x;
  if (i < n) out[i] = (_Float16)in[i];
}

__global__ void neg_exp_k(const float* __restrict__ in, float* __restrict__ out, int n) {
  int i = blockIdx.x * blockDim.x + threadIdx.x;
  if (i < n) out[i] = -__expf(in[i]);
}

// ---------------------------------------------------------------- tiled GEMM
// C[M,N] = A[M,K] * W[N,K]^T. 256 threads = 8 waves; macro tile 64(M)x128(N);
// wave (wm,wn) owns 32x32 = 2x2 WMMA tiles. K stepped by 32 through LDS.
#define BM 64
#define BN 128
#define BK 32

__global__ void __launch_bounds__(256) gemm_wmma_tiled_k(
    const _Float16* __restrict__ A, const _Float16* __restrict__ W,
    float* __restrict__ C, int M, int N, int K) {
  __shared__ _Float16 sA[BM * BK];   // 4 KB
  __shared__ _Float16 sB[BN * BK];   // 8 KB
  const int tid  = threadIdx.x;
  const int lane = tid & 31;
  const int wave = tid >> 5;           // 0..7
  const int wm   = (wave >> 2) * 32;   // 0 / 32
  const int wn   = (wave & 3) * 32;    // 0 / 32 / 64 / 96
  const int g    = lane >> 4;
  const int mr   = lane & 15;

  const int tile_m = blockIdx.y * BM;
  const int tile_n = blockIdx.x * BN;

  // staging map: thread -> (row, 8-half chunk)
  const int sr = tid >> 2;           // 0..63
  const int sc = (tid & 3) * 8;      // 0,8,16,24

  v8f acc00 = {}, acc01 = {}, acc10 = {}, acc11 = {};

  for (int kk = 0; kk < K; kk += BK) {
    __syncthreads();   // protect previous iteration's fragment reads
    const _Float16* ga  = A + (size_t)(tile_m + sr) * K + kk + sc;
    const _Float16* gb0 = W + (size_t)(tile_n + sr) * K + kk + sc;
    const _Float16* gb1 = W + (size_t)(tile_n + sr + 64) * K + kk + sc;
#ifdef USE_ASYNC_LDS
    __builtin_amdgcn_global_load_async_to_lds_b128(
        (gptr_t)ga,  (lptr_t)(sA + sr * BK + sc), 0, 0);
    __builtin_amdgcn_global_load_async_to_lds_b128(
        (gptr_t)gb0, (lptr_t)(sB + sr * BK + sc), 0, 0);
    __builtin_amdgcn_global_load_async_to_lds_b128(
        (gptr_t)gb1, (lptr_t)(sB + (sr + 64) * BK + sc), 0, 0);
    __builtin_amdgcn_s_wait_asynccnt(0);
#else
    *(v8h*)(sA + sr * BK + sc)        = *(const v8h*)ga;
    *(v8h*)(sB + sr * BK + sc)        = *(const v8h*)gb0;
    *(v8h*)(sB + (sr + 64) * BK + sc) = *(const v8h*)gb1;
#endif
    __syncthreads();

    v16h a0 = ldfrag(sA + (wm + mr) * BK + g * 8);
    v16h a1 = ldfrag(sA + (wm + 16 + mr) * BK + g * 8);
    v16h b0 = ldfrag(sB + (wn + mr) * BK + g * 8);
    v16h b1 = ldfrag(sB + (wn + 16 + mr) * BK + g * 8);

    acc00 = wmma_step(a0, b0, acc00);
    acc01 = wmma_step(a0, b1, acc01);
    acc10 = wmma_step(a1, b0, acc10);
    acc11 = wmma_step(a1, b1, acc11);
  }

  // C layout per tile: VGPR r -> row = r + g*8, col = mr
  const int row0 = tile_m + wm + g * 8;
  const int col0 = tile_n + wn + mr;
  float* c00 = C + (size_t)row0 * N + col0;
#pragma unroll
  for (int r = 0; r < 8; ++r) {
    c00[(size_t)r * N]             = acc00[r];
    c00[(size_t)r * N + 16]        = acc01[r];
    c00[(size_t)(r + 16) * N]      = acc10[r];
    c00[(size_t)(r + 16) * N + 16] = acc11[r];
  }
}

// ---------------------------------------------------------------- small GEMM
// one wave per 16x16 tile (for x_proj, N=64)
__global__ void __launch_bounds__(32) gemm_wmma_f16_k(
    const _Float16* __restrict__ A, const _Float16* __restrict__ W,
    float* __restrict__ C, int M, int N, int K) {
  const int lane = threadIdx.x & 31;
  const int g    = lane >> 4;
  const int mr   = lane & 15;
  const int tile_n = blockIdx.x * 16;
  const int tile_m = blockIdx.y * 16;

  const _Float16* arow = A + (size_t)(tile_m + mr) * K;
  const _Float16* brow = W + (size_t)(tile_n + mr) * K;

  v8f acc = {};
  for (int kk = 0; kk < K; kk += 32) {
    v16h a = ldfrag(arow + kk + g * 8);
    v16h b = ldfrag(brow + kk + g * 8);
    acc = wmma_step(a, b, acc);
  }
  float* crow = C + (size_t)(tile_m + g * 8) * N + tile_n + mr;
#pragma unroll
  for (int r = 0; r < 8; ++r) crow[(size_t)r * N] = acc[r];
}

// ---------------------------------------------------------------- conv+silu
__global__ void conv_silu_k(const float* __restrict__ xz,
                            const float* __restrict__ conv_w,
                            const float* __restrict__ conv_b,
                            float* __restrict__ u, _Float16* __restrict__ uh) {
  int idx = blockIdx.x * blockDim.x + threadIdx.x;
  if (idx >= MROWS * DINNER) return;
  int d = idx % DINNER;
  int m = idx / DINNER;          // m = b*SEQ + l
  int l = m % SEQ;
  float acc = conv_b[d];
#pragma unroll
  for (int j = 0; j < DCONV; ++j) {
    int ls = l - (DCONV - 1) + j;
    if (ls >= 0)
      acc += conv_w[d * DCONV + j] * xz[(size_t)(m - (DCONV - 1) + j) * (2 * DINNER) + d];
  }
  float s = acc * sigmoidf_(acc);
  u[idx]  = s;
  uh[idx] = (_Float16)s;
}

__global__ void pack_dt_k(const float* __restrict__ x_dbl, _Float16* __restrict__ dth) {
  int idx = blockIdx.x * blockDim.x + threadIdx.x;
  if (idx >= MROWS * DTRANK) return;
  int m = idx / DTRANK, r = idx % DTRANK;
  dth[idx] = (_Float16)x_dbl[(size_t)m * (DTRANK + 2 * DSTATE) + r];
}

__global__ void softplus_bias_k(const float* __restrict__ dpre,
                                const float* __restrict__ bias,
                                float* __restrict__ delta) {
  int idx = blockIdx.x * blockDim.x + threadIdx.x;
  if (idx >= MROWS * DINNER) return;
  float v = dpre[idx] + bias[idx % DINNER];
  delta[idx] = (v > 20.f) ? v : log1pf(__expf(v));
}

// ---------------------------------------------------------------- scan
__global__ void __launch_bounds__(256) selective_scan_k(
    const float* __restrict__ u, const float* __restrict__ delta,
    const float* __restrict__ Aneg, const float* __restrict__ x_dbl,
    const float* __restrict__ Dp, float* __restrict__ y) {
  const int d = blockIdx.x * 256 + threadIdx.x;
  const int b = blockIdx.y;
  __shared__ float sB[16 * DSTATE];
  __shared__ float sC[16 * DSTATE];

  float h[DSTATE];
#pragma unroll
  for (int n = 0; n < DSTATE; ++n) h[n] = 0.f;
  float Ad[DSTATE];
#pragma unroll
  for (int n = 0; n < DSTATE; ++n) Ad[n] = Aneg[d * DSTATE + n];
  const float Dd = Dp[d];

  for (int l0 = 0; l0 < SEQ; l0 += 16) {
    __syncthreads();
    for (int i = threadIdx.x; i < 16 * 32; i += 256) {
      int lr = i >> 5;
      int c  = i & 31;
      float v = x_dbl[(size_t)(b * SEQ + l0 + lr) * (DTRANK + 2 * DSTATE) + DTRANK + c];
      if (c < DSTATE) sB[lr * DSTATE + c] = v;
      else            sC[lr * DSTATE + (c - DSTATE)] = v;
    }
    __syncthreads();
#pragma unroll 4
    for (int li = 0; li < 16; ++li) {
      int m = b * SEQ + l0 + li;
      float dl = delta[(size_t)m * DINNER + d];
      float uu = u[(size_t)m * DINNER + d];
      float du = dl * uu;
      float acc = 0.f;
#pragma unroll
      for (int n = 0; n < DSTATE; ++n) {
        float dA = __expf(dl * Ad[n]);
        h[n] = dA * h[n] + du * sB[li * DSTATE + n];
        acc += h[n] * sC[li * DSTATE + n];
      }
      y[(size_t)m * DINNER + d] = acc + uu * Dd;
    }
  }
}

__global__ void gate_k(const float* __restrict__ y, const float* __restrict__ xz,
                       _Float16* __restrict__ yh) {
  int idx = blockIdx.x * blockDim.x + threadIdx.x;
  if (idx >= MROWS * DINNER) return;
  int m = idx / DINNER, d = idx % DINNER;
  float z = xz[(size_t)m * (2 * DINNER) + DINNER + d];
  yh[idx] = (_Float16)(y[idx] * (z * sigmoidf_(z)));
}

__global__ void __launch_bounds__(256) residual_ln_k(
    const float* __restrict__ x, const float* __restrict__ ssm,
    const float* __restrict__ w, const float* __restrict__ bias,
    float* __restrict__ out) {
  const int m = blockIdx.x;
  const int t = threadIdx.x;
  __shared__ float red[256];
  __shared__ float s_mu, s_inv;
  const size_t base = (size_t)m * DMODEL;
  float r0 = x[base + t]       + ssm[base + t];
  float r1 = x[base + t + 256] + ssm[base + t + 256];
  red[t] = r0 + r1;
  __syncthreads();
  for (int s = 128; s > 0; s >>= 1) { if (t < s) red[t] += red[t + s]; __syncthreads(); }
  if (t == 0) s_mu = red[0] / (float)DMODEL;
  __syncthreads();
  float mu = s_mu;
  float d0 = r0 - mu, d1 = r1 - mu;
  red[t] = d0 * d0 + d1 * d1;
  __syncthreads();
  for (int s = 128; s > 0; s >>= 1) { if (t < s) red[t] += red[t + s]; __syncthreads(); }
  if (t == 0) s_inv = rsqrtf(red[0] / (float)DMODEL + 1e-5f);
  __syncthreads();
  float inv = s_inv;
  out[base + t]       = d0 * inv * w[t]       + bias[t];
  out[base + t + 256] = d1 * inv * w[t + 256] + bias[t + 256];
}

// ---------------------------------------------------------------- launch
static void* ws_take(char*& p, size_t bytes) {
  void* r = (void*)p;
  p += (bytes + 255) & ~(size_t)255;
  return r;
}

extern "C" void kernel_launch(void* const* d_in, const int* in_sizes, int n_in,
                              void* d_out, int out_size, void* d_ws, size_t ws_size,
                              hipStream_t stream) {
  (void)in_sizes; (void)n_in; (void)out_size; (void)ws_size;
  const float* x          = (const float*)d_in[0];
  const float* in_proj_w  = (const float*)d_in[1];
  const float* conv_w     = (const float*)d_in[2];
  const float* conv_b     = (const float*)d_in[3];
  const float* x_proj_w   = (const float*)d_in[4];
  const float* dt_proj_w  = (const float*)d_in[5];
  const float* dt_proj_b  = (const float*)d_in[6];
  const float* A_log      = (const float*)d_in[7];
  const float* Dp         = (const float*)d_in[8];
  const float* out_proj_w = (const float*)d_in[9];
  const float* ln_w       = (const float*)d_in[10];
  const float* ln_b       = (const float*)d_in[11];
  float* out = (float*)d_out;

  char* wp = (char*)d_ws;
  _Float16* x_h      = (_Float16*)ws_take(wp, (size_t)MROWS * DMODEL * 2);
  _Float16* w_in_h   = (_Float16*)ws_take(wp, (size_t)2 * DINNER * DMODEL * 2);
  _Float16* w_xp_h   = (_Float16*)ws_take(wp, (size_t)(DTRANK + 2 * DSTATE) * DINNER * 2);
  _Float16* w_dt_h   = (_Float16*)ws_take(wp, (size_t)DINNER * DTRANK * 2);
  _Float16* w_out_h  = (_Float16*)ws_take(wp, (size_t)DMODEL * DINNER * 2);
  float*    Aneg     = (float*)   ws_take(wp, (size_t)DINNER * DSTATE * 4);
  float*    xz       = (float*)   ws_take(wp, (size_t)MROWS * 2 * DINNER * 4);
  float*    u_f32    = (float*)   ws_take(wp, (size_t)MROWS * DINNER * 4);
  _Float16* u_h      = (_Float16*)ws_take(wp, (size_t)MROWS * DINNER * 2);
  float*    x_dbl    = (float*)   ws_take(wp, (size_t)MROWS * (DTRANK + 2 * DSTATE) * 4);
  _Float16* dt_h     = (_Float16*)ws_take(wp, (size_t)MROWS * DTRANK * 2);
  float*    delta    = (float*)   ws_take(wp, (size_t)MROWS * DINNER * 4);
  float*    y_f32    = (float*)   ws_take(wp, (size_t)MROWS * DINNER * 4);
  _Float16* y_h      = (_Float16*)ws_take(wp, (size_t)MROWS * DINNER * 2);
  float*    ssm      = (float*)   ws_take(wp, (size_t)MROWS * DMODEL * 4);

  const int T = 256;
  auto blks = [](size_t n, int t) { return (int)((n + t - 1) / t); };

  cvt_f32_to_f16_k<<<blks((size_t)MROWS * DMODEL, T), T, 0, stream>>>(x, x_h, MROWS * DMODEL);
  cvt_f32_to_f16_k<<<blks((size_t)2 * DINNER * DMODEL, T), T, 0, stream>>>(in_proj_w, w_in_h, 2 * DINNER * DMODEL);
  cvt_f32_to_f16_k<<<blks((size_t)(DTRANK + 2 * DSTATE) * DINNER, T), T, 0, stream>>>(x_proj_w, w_xp_h, (DTRANK + 2 * DSTATE) * DINNER);
  cvt_f32_to_f16_k<<<blks((size_t)DINNER * DTRANK, T), T, 0, stream>>>(dt_proj_w, w_dt_h, DINNER * DTRANK);
  cvt_f32_to_f16_k<<<blks((size_t)DMODEL * DINNER, T), T, 0, stream>>>(out_proj_w, w_out_h, DMODEL * DINNER);
  neg_exp_k<<<blks((size_t)DINNER * DSTATE, T), T, 0, stream>>>(A_log, Aneg, DINNER * DSTATE);

  // in_proj: xz[4096,2048] = x_h . in_proj_w^T   (tiled WMMA)
  gemm_wmma_tiled_k<<<dim3((2 * DINNER) / BN, MROWS / BM), 256, 0, stream>>>(
      x_h, w_in_h, xz, MROWS, 2 * DINNER, DMODEL);

  conv_silu_k<<<blks((size_t)MROWS * DINNER, T), T, 0, stream>>>(xz, conv_w, conv_b, u_f32, u_h);

  // x_proj: x_dbl[4096,64] (N too narrow for macro tile -> per-wave kernel)
  gemm_wmma_f16_k<<<dim3((DTRANK + 2 * DSTATE) / 16, MROWS / 16), 32, 0, stream>>>(
      u_h, w_xp_h, x_dbl, MROWS, DTRANK + 2 * DSTATE, DINNER);

  // dt_proj: delta_pre[4096,1024] = dt[4096,32] . dt_proj_w^T (tiled, K=32)
  pack_dt_k<<<blks((size_t)MROWS * DTRANK, T), T, 0, stream>>>(x_dbl, dt_h);
  gemm_wmma_tiled_k<<<dim3(DINNER / BN, MROWS / BM), 256, 0, stream>>>(
      dt_h, w_dt_h, delta, MROWS, DINNER, DTRANK);
  softplus_bias_k<<<blks((size_t)MROWS * DINNER, T), T, 0, stream>>>(delta, dt_proj_b, delta);

  selective_scan_k<<<dim3(DINNER / 256, BATCH), 256, 0, stream>>>(
      u_f32, delta, Aneg, x_dbl, Dp, y_f32);

  gate_k<<<blks((size_t)MROWS * DINNER, T), T, 0, stream>>>(y_f32, xz, y_h);

  // out_proj: ssm[4096,512] = y . out_proj_w^T (tiled WMMA)
  gemm_wmma_tiled_k<<<dim3(DMODEL / BN, MROWS / BM), 256, 0, stream>>>(
      y_h, w_out_h, ssm, MROWS, DMODEL, DINNER);

  residual_ln_k<<<MROWS, 256, 0, stream>>>(x, ssm, ln_w, ln_b, out);
}
